// pasep_29970281792198
// MI455X (gfx1250) — matrix operand order repaired and checked by reading, not compiled, not run on hardware
//
#include <hip/hip_runtime.h>
#include <hip/hip_bf16.h>
#include <math.h>

typedef __attribute__((ext_vector_type(16))) _Float16 v16h;
typedef __attribute__((ext_vector_type(8)))  _Float16 v8h;
typedef __attribute__((ext_vector_type(8)))  float    v8f;

#define PI_F 3.14159265358979f

// ---------------------------------------------------------------- helpers

__device__ __forceinline__ int reflect_idx(int i, int T) {
  if (i < 0) i = -i;
  if (i >= T) i = 2 * T - 2 - i;
  return i;
}

// ---------------------------------------------------------------- weight preparation

// Sinc filter -> shuffled f16 B-layout [chunk][ntile=4][lane][16], KKpad=256, N=64
__global__ void pasep_sinc_filter(const float* __restrict__ low_hz,
                                  const float* __restrict__ band_hz,
                                  _Float16* __restrict__ dst) {
  int c = blockIdx.x;     // channel n, 0..63
  int k = threadIdx.x;    // tap kk, 0..255 (padded)
  float low  = 50.0f + fabsf(low_hz[c]);
  float high = low + 50.0f + fabsf(band_hz[c]);
  high = fminf(fmaxf(high, 50.0f), 8000.0f);
  float band = high - low;
  float val = 0.0f;
  if (k < 251) {
    if (k == 125) {
      val = 1.0f;  // 2*band / (2*band)
    } else {
      int i = (k < 125) ? k : (250 - k);
      float n = 2.0f * PI_F * (float)(i - 125) / 16000.0f;
      float wini = (float)i * (124.5f / 124.0f);
      float window = 0.54f - 0.46f * cosf(2.0f * PI_F * wini / 251.0f);
      float left = (sinf(high * n) - sinf(low * n)) / (n * 0.5f) * window;
      val = left / (2.0f * band);
    }
  }
  int chunk = k >> 5;
  int lane  = (((k & 31) >= 16) ? 16 : 0) + (c & 15);
  int p     = k & 15;
  int tile  = c >> 4;
  dst[(((chunk * 4 + tile) * 32) + lane) * 16 + p] = (_Float16)val;
}

// f32 weights [KK][N] -> shuffled f16 [KKpad/32][N/16][lane=32][16], zero padding rows
__global__ void pasep_shuffle_w(const float* __restrict__ src, _Float16* __restrict__ dst,
                                int KK, int KKpad, int N) {
  int idx = blockIdx.x * blockDim.x + threadIdx.x;
  if (idx >= KKpad * N) return;
  int p    = idx & 15;
  int lane = (idx >> 4) & 31;
  int rest = idx >> 9;                 // chunk * ntiles + tile
  int ntiles = N >> 4;
  int tile  = rest % ntiles;
  int chunk = rest / ntiles;
  int kk = chunk * 32 + ((lane >> 4) << 4) + p;
  int n  = tile * 16 + (lane & 15);
  dst[idx] = (kk < KK) ? (_Float16)src[(long long)kk * N + n] : (_Float16)0.0f;
}

// reflect-pad f32 [B,T,C] -> f16 [B,TpadA,C]; rows >= Tpad zero-filled (slack for kk tail)
__global__ void pasep_pad_reflect(const float* __restrict__ src, _Float16* __restrict__ dst,
                                  int B, int T, int C, int pl, int Tpad, int TpadA) {
  int idx = blockIdx.x * blockDim.x + threadIdx.x;
  int total = B * TpadA * C;
  if (idx >= total) return;
  int c = idx % C;
  int r = idx / C;
  int tp = r % TpadA;
  int b = r / TpadA;
  if (tp >= Tpad) { dst[idx] = (_Float16)0.0f; return; }
  int t = reflect_idx(tp - pl, T);
  dst[idx] = (_Float16)src[((long long)b * T + t) * C + c];
}

// ---------------------------------------------------------------- conv (implicit GEMM, WMMA)
// M-tile 32 x N-tile 64 (4 waves x 16). A staged in LDS (fragment-native order,
// double-buffered, 1 barrier/chunk); B from pre-shuffled weights (32B/lane).

template <bool CIN1>
__global__ __launch_bounds__(128)
void pasep_conv_bn_prelu(const _Float16* __restrict__ xpad, const _Float16* __restrict__ wshuf,
                         const float* __restrict__ bias,
                         const float* __restrict__ bn_mean, const float* __restrict__ bn_var,
                         const float* __restrict__ bn_gamma, const float* __restrict__ bn_beta,
                         const float* __restrict__ alpha,
                         float* __restrict__ out,
                         int TpadA, int Cin, int Cout, int Tout, int stride, int KKpad) {
  __shared__ __align__(32) _Float16 ldsA[2048];   // 2 buffers x (2 subtiles x 32 x 16)
  const int tid = threadIdx.x;
  const int lane = tid & 31;
  const int wave = tid >> 5;
  const int t0 = blockIdx.x * 32;
  const int n0w = blockIdx.y * 64 + wave * 16;
  const int b = blockIdx.z;
  const _Float16* xb = xpad + (size_t)b * TpadA * Cin;

  // cooperative A loader: thread -> (row lm, kkl base lk), 8 contiguous halfs
  const int lm = tid >> 2;              // 0..31
  const int lk = (tid & 3) * 8;         // 0,8,16,24
  int trow = t0 + lm;
  if (trow > Tout - 1) trow = Tout - 1; // clamp partial tiles (masked at store)
  const _Float16* grow = xb + (size_t)trow * stride * Cin + lk;   // addr linear in kk
  const int slot = ((lm >> 4) ? 512 : 0) + ((lm & 15) + ((lk & 16) ? 16 : 0)) * 16 +
                   ((lk & 7) + ((lk & 8) ? 8 : 0));

  // pre-shuffled B: [chunk][Cout/16][lane][16]
  const size_t bchunk = (size_t)(Cout >> 4) * 32 * 16;
  const _Float16* bp = wshuf + ((size_t)(n0w >> 4) * 32 + lane) * 16;

  auto stage = [&](const _Float16* p) -> v8h {
    if constexpr (CIN1) {
      v8h g;
#pragma unroll
      for (int u = 0; u < 8; ++u) g[u] = p[u];
      return g;
    } else {
      return *(const v8h*)p;
    }
  };

  // prologue: chunk 0 into buffer 0
  *(v8h*)&ldsA[slot] = stage(grow);
  __syncthreads();

  v8f acc[2] = {};
  int cur = 0;
  int kk0 = 0;
  for (; kk0 < KKpad - 32; kk0 += 32) {
    v8h gn = stage(grow + kk0 + 32);              // next chunk in flight
    const _Float16* bufc = &ldsA[cur << 10];
    v16h a0 = *(const v16h*)&bufc[lane * 16];
    v16h a1 = *(const v16h*)&bufc[512 + lane * 16];
    v16h bf = *(const v16h*)(bp + (size_t)(kk0 >> 5) * bchunk);
    acc[0] = __builtin_amdgcn_wmma_f32_16x16x32_f16(false, a0, false, bf, (short)0, acc[0],
                                                    false, false);
    acc[1] = __builtin_amdgcn_wmma_f32_16x16x32_f16(false, a1, false, bf, (short)0, acc[1],
                                                    false, false);
    *(v8h*)&ldsA[((cur ^ 1) << 10) + slot] = gn;  // fill other buffer
    __syncthreads();
    cur ^= 1;
  }
  {  // peeled last chunk
    const _Float16* bufc = &ldsA[cur << 10];
    v16h a0 = *(const v16h*)&bufc[lane * 16];
    v16h a1 = *(const v16h*)&bufc[512 + lane * 16];
    v16h bf = *(const v16h*)(bp + (size_t)(kk0 >> 5) * bchunk);
    acc[0] = __builtin_amdgcn_wmma_f32_16x16x32_f16(false, a0, false, bf, (short)0, acc[0],
                                                    false, false);
    acc[1] = __builtin_amdgcn_wmma_f32_16x16x32_f16(false, a1, false, bf, (short)0, acc[1],
                                                    false, false);
  }

  // fused bias + batchnorm + PReLU epilogue
  int n = lane & 15;
  int mh = (lane >> 4) << 3;
  int co = n0w + n;
  float bi = bias ? bias[co] : 0.0f;
  float g  = bn_gamma[co] * rsqrtf(bn_var[co] + 1e-5f);
  float mu = bn_mean[co], be = bn_beta[co], al = alpha[co];
#pragma unroll
  for (int s = 0; s < 2; ++s) {
#pragma unroll
    for (int r = 0; r < 8; ++r) {
      int t = t0 + s * 16 + r + mh;
      if (t < Tout) {
        float v = acc[s][r] + bi;
        v = (v - mu) * g + be;
        v = (v >= 0.0f) ? v : al * v;
        out[((size_t)b * Tout + t) * Cout + co] = v;
      }
    }
  }
}

// ---------------------------------------------------------------- generic GEMM (WMMA)
// out[M,N] f32 = A[M,KKpad] f16 x shuffled-B + bias (+= if accumulate). M % 32 == 0.

__global__ __launch_bounds__(128)
void pasep_gemm(const _Float16* __restrict__ A, const _Float16* __restrict__ wshuf,
                const float* __restrict__ bias, float* __restrict__ out,
                int N, int KKpad, int accumulate) {
  __shared__ __align__(32) _Float16 ldsA[2048];
  const int tid = threadIdx.x;
  const int lane = tid & 31;
  const int wave = tid >> 5;
  const int m0 = blockIdx.x * 32;
  const int n0w = blockIdx.y * 64 + wave * 16;

  const int lm = tid >> 2;
  const int lk = (tid & 3) * 8;
  const _Float16* grow = A + (size_t)(m0 + lm) * KKpad + lk;
  const int slot = ((lm >> 4) ? 512 : 0) + ((lm & 15) + ((lk & 16) ? 16 : 0)) * 16 +
                   ((lk & 7) + ((lk & 8) ? 8 : 0));

  const size_t bchunk = (size_t)(N >> 4) * 32 * 16;
  const _Float16* bp = wshuf + ((size_t)(n0w >> 4) * 32 + lane) * 16;

  *(v8h*)&ldsA[slot] = *(const v8h*)grow;
  __syncthreads();

  v8f acc[2] = {};
  int cur = 0;
  int kk0 = 0;
  for (; kk0 < KKpad - 32; kk0 += 32) {
    v8h gn = *(const v8h*)(grow + kk0 + 32);
    const _Float16* bufc = &ldsA[cur << 10];
    v16h a0 = *(const v16h*)&bufc[lane * 16];
    v16h a1 = *(const v16h*)&bufc[512 + lane * 16];
    v16h bf = *(const v16h*)(bp + (size_t)(kk0 >> 5) * bchunk);
    acc[0] = __builtin_amdgcn_wmma_f32_16x16x32_f16(false, a0, false, bf, (short)0, acc[0],
                                                    false, false);
    acc[1] = __builtin_amdgcn_wmma_f32_16x16x32_f16(false, a1, false, bf, (short)0, acc[1],
                                                    false, false);
    *(v8h*)&ldsA[((cur ^ 1) << 10) + slot] = gn;
    __syncthreads();
    cur ^= 1;
  }
  {
    const _Float16* bufc = &ldsA[cur << 10];
    v16h a0 = *(const v16h*)&bufc[lane * 16];
    v16h a1 = *(const v16h*)&bufc[512 + lane * 16];
    v16h bf = *(const v16h*)(bp + (size_t)(kk0 >> 5) * bchunk);
    acc[0] = __builtin_amdgcn_wmma_f32_16x16x32_f16(false, a0, false, bf, (short)0, acc[0],
                                                    false, false);
    acc[1] = __builtin_amdgcn_wmma_f32_16x16x32_f16(false, a1, false, bf, (short)0, acc[1],
                                                    false, false);
  }

  int n = n0w + (lane & 15);
  int mh = (lane >> 4) << 3;
  float bi = bias ? bias[n] : 0.0f;
#pragma unroll
  for (int s = 0; s < 2; ++s) {
#pragma unroll
    for (int r = 0; r < 8; ++r) {
      int m = m0 + s * 16 + r + mh;
      size_t oi = (size_t)m * N + n;
      float v = acc[s][r] + bi;
      if (accumulate) v += out[oi];
      out[oi] = v;
    }
  }
}

// ---------------------------------------------------------------- pooling / QRNN glue

__global__ void pasep_avgpool(const float* __restrict__ src, _Float16* __restrict__ dst,
                              int T, int C, int factor) {
  int idx = blockIdx.x * blockDim.x + threadIdx.x;
  int total = 8 * 200 * C;
  if (idx >= total) return;
  int c = idx % C;
  int r = idx / C;           // b*200 + t
  int t = r % 200;
  int b = r / 200;
  const float* p = src + ((size_t)b * T + (size_t)t * factor) * C + c;
  float s = 0.0f;
  for (int j = 0; j < factor; ++j) s += p[(size_t)j * C];
  dst[idx] = (_Float16)(s / (float)factor);
}

__global__ void pasep_build_qin(const float* __restrict__ h7, _Float16* __restrict__ qin) {
  int idx = blockIdx.x * blockDim.x + threadIdx.x;
  if (idx >= 1600 * 1024) return;
  int c = idx & 1023;
  int r = idx >> 10;
  int t = r % 200;
  int b = r / 200;
  float v;
  if (c < 512) v = h7[((size_t)b * 200 + t) * 512 + c];
  else v = (t > 0) ? h7[((size_t)b * 200 + t - 1) * 512 + (c - 512)] : 0.0f;
  qin[idx] = (_Float16)v;
}

__global__ void pasep_qrnn_scan(const float* __restrict__ g, _Float16* __restrict__ hq) {
  int b = blockIdx.x;     // 0..7
  int d = threadIdx.x;    // 0..511
  float c = 0.0f;
  for (int t = 0; t < 200; ++t) {
    const float* gr = g + (size_t)(b * 200 + t) * 1536;
    float z = tanhf(gr[d]);
    float f = 1.0f / (1.0f + expf(-gr[512 + d]));
    float o = 1.0f / (1.0f + expf(-gr[1024 + d]));
    c = f * z + (1.0f - f) * c;
    hq[(size_t)(b * 200 + t) * 512 + d] = (_Float16)(o * c);
  }
}

__global__ void pasep_final_norm(const float* __restrict__ acc, const float* __restrict__ mean,
                                 const float* __restrict__ var, float* __restrict__ out) {
  int idx = blockIdx.x * blockDim.x + threadIdx.x;
  if (idx >= 1600 * 256) return;
  int c = idx & 255;
  out[idx] = (acc[idx] - mean[c]) * rsqrtf(var[c] + 1e-5f);
}

// ---------------------------------------------------------------- host orchestration

extern "C" void kernel_launch(void* const* d_in, const int* in_sizes, int n_in,
                              void* d_out, int out_size, void* d_ws, size_t ws_size,
                              hipStream_t stream) {
  (void)in_sizes; (void)n_in; (void)out_size; (void)ws_size;

  const int Tin [8] = {32000, 32000, 3200, 1600, 1600, 800, 800, 400};
  const int Cin [8] = {1, 64, 64, 128, 128, 256, 256, 512};
  const int Cout[8] = {64, 64, 128, 128, 256, 256, 512, 512};
  const int Kk  [8] = {251, 20, 11, 11, 11, 11, 11, 11};
  const int St  [8] = {1, 10, 2, 1, 2, 1, 2, 2};
  const int PL  [8] = {125, 9, 4, 5, 4, 5, 4, 4};
  const int Tpad[8] = {32250, 32019, 3209, 1610, 1609, 810, 809, 409};
  const int Tout[8] = {32000, 3200, 1600, 1600, 800, 800, 400, 200};
  int KK[8], KKp[8], TpadA[8];
  for (int i = 0; i < 8; ++i) {
    KK[i] = Kk[i] * Cin[i];
    KKp[i] = (KK[i] + 31) & ~31;
    TpadA[i] = Tpad[i] + 32;     // zero-filled slack rows for kk-tail reads
  }

  const float* x = (const float*)d_in[0];
  const float* sinc_low  = (const float*)d_in[1];
  const float* sinc_band = (const float*)d_in[2];
  const float* qrnn_w = (const float*)d_in[64];
  const float* qrnn_b = (const float*)d_in[65];
  const float* out_w  = (const float*)d_in[66];
  const float* out_b  = (const float*)d_in[67];
  const float* norm_mean = (const float*)d_in[68];
  const float* norm_var  = (const float*)d_in[69];

  // d_out layout: out, fe1, fe2, fe4, fe6, fe7
  float* outp = (float*)d_out;
  float* fe1 = outp + 8 * 200 * 256;
  float* fe2 = fe1 + (size_t)8 * 32000 * 64;
  float* fe4 = fe2 + (size_t)8 * 3200 * 64;
  float* fe6 = fe4 + (size_t)8 * 1600 * 128;
  float* fe7 = fe6 + (size_t)8 * 800 * 256;

  char* base = (char*)d_ws;
  size_t off = 0;
  auto alloc = [&](size_t bytes) -> void* {
    void* p = base + off;
    off = (off + bytes + 255) & ~(size_t)255;
    return p;
  };

  _Float16* w16[8];
  for (int i = 0; i < 8; ++i) w16[i] = (_Float16*)alloc((size_t)KKp[i] * Cout[i] * 2);
  const int DCin[7] = {64, 64, 128, 128, 256, 256, 512};
  _Float16* dw16[7];
  for (int i = 0; i < 7; ++i) dw16[i] = (_Float16*)alloc((size_t)DCin[i] * 256 * 2);
  _Float16* qw16 = (_Float16*)alloc((size_t)1024 * 1536 * 2);
  _Float16* ow16 = (_Float16*)alloc((size_t)512 * 256 * 2);
  _Float16* padb[8];
  for (int i = 0; i < 8; ++i) padb[i] = (_Float16*)alloc((size_t)8 * TpadA[i] * Cin[i] * 2);
  float* b2f = (float*)alloc((size_t)8 * 1600 * 128 * 4);
  float* b4f = (float*)alloc((size_t)8 * 800 * 256 * 4);
  float* h7f = (float*)alloc((size_t)8 * 200 * 512 * 4);
  _Float16* hp[7];
  for (int i = 0; i < 7; ++i) hp[i] = (_Float16*)alloc((size_t)1600 * DCin[i] * 2);
  _Float16* qin = (_Float16*)alloc((size_t)1600 * 1024 * 2);
  float* gbuf = (float*)alloc((size_t)1600 * 1536 * 4);
  float* accb = (float*)alloc((size_t)1600 * 256 * 4);
  _Float16* hq = (_Float16*)alloc((size_t)1600 * 512 * 2);

  float* act[8] = {fe1, fe2, b2f, fe4, b4f, fe6, fe7, h7f};

  // ---- 1. weight preparation (shuffled into WMMA B-fragment order)
  pasep_sinc_filter<<<64, 256, 0, stream>>>(sinc_low, sinc_band, w16[0]);
  for (int i = 1; i < 8; ++i) {
    const float* ksrc = (const float*)d_in[7 * i + 6];
    int total = KKp[i] * Cout[i];
    pasep_shuffle_w<<<(total + 255) / 256, 256, 0, stream>>>(ksrc, w16[i], KK[i], KKp[i],
                                                             Cout[i]);
  }
  for (int i = 0; i < 7; ++i) {
    const float* dsrc = (const float*)d_in[57 + i];
    int total = DCin[i] * 256;
    pasep_shuffle_w<<<(total + 255) / 256, 256, 0, stream>>>(dsrc, dw16[i], DCin[i], DCin[i],
                                                             256);
  }
  pasep_shuffle_w<<<(1024 * 1536 + 255) / 256, 256, 0, stream>>>(qrnn_w, qw16, 1024, 1024, 1536);
  pasep_shuffle_w<<<(512 * 256 + 255) / 256, 256, 0, stream>>>(out_w, ow16, 512, 512, 256);

  // ---- 2. conv stack
  for (int i = 0; i < 8; ++i) {
    const float* src = (i == 0) ? x : act[i - 1];
    int padtot = 8 * TpadA[i] * Cin[i];
    pasep_pad_reflect<<<(padtot + 255) / 256, 256, 0, stream>>>(src, padb[i], 8, Tin[i],
                                                                Cin[i], PL[i], Tpad[i],
                                                                TpadA[i]);
    int bb = (i == 0) ? 3 : 7 * i + 1;
    const float* bias = (i == 0) ? nullptr : (const float*)d_in[7 * i + 7];
    dim3 grid((Tout[i] + 31) / 32, Cout[i] / 64, 8);
    if (i == 0) {
      pasep_conv_bn_prelu<true><<<grid, 128, 0, stream>>>(
          padb[i], w16[i], bias,
          (const float*)d_in[bb + 0], (const float*)d_in[bb + 1],
          (const float*)d_in[bb + 2], (const float*)d_in[bb + 3],
          (const float*)d_in[bb + 4],
          act[i], TpadA[i], Cin[i], Cout[i], Tout[i], St[i], KKp[i]);
    } else {
      pasep_conv_bn_prelu<false><<<grid, 128, 0, stream>>>(
          padb[i], w16[i], bias,
          (const float*)d_in[bb + 0], (const float*)d_in[bb + 1],
          (const float*)d_in[bb + 2], (const float*)d_in[bb + 3],
          (const float*)d_in[bb + 4],
          act[i], TpadA[i], Cin[i], Cout[i], Tout[i], St[i], KKp[i]);
    }
  }

  // ---- 3. QRNN
  pasep_build_qin<<<(1600 * 1024 + 255) / 256, 256, 0, stream>>>(h7f, qin);
  pasep_gemm<<<dim3(50, 24), 128, 0, stream>>>(qin, qw16, qrnn_b, gbuf, 1536, 1024, 0);
  pasep_qrnn_scan<<<8, 512, 0, stream>>>(gbuf, hq);

  // ---- 4. output projection + dense skip sums (pool-before-GEMM)
  pasep_gemm<<<dim3(50, 4), 128, 0, stream>>>(hq, ow16, out_b, accb, 256, 512, 0);
  const int Tact[7] = {32000, 3200, 1600, 1600, 800, 800, 400};
  for (int i = 0; i < 7; ++i) {
    int factor = Tact[i] / 200;
    int total = 1600 * DCin[i];
    pasep_avgpool<<<(total + 255) / 256, 256, 0, stream>>>(act[i], hp[i], Tact[i], DCin[i],
                                                           factor);
    pasep_gemm<<<dim3(50, 4), 128, 0, stream>>>(hp[i], dw16[i], nullptr, accb, 256, DCin[i],
                                                1);
  }

  // ---- 5. final normalization
  pasep_final_norm<<<(1600 * 256 + 255) / 256, 256, 0, stream>>>(accb, norm_mean, norm_var,
                                                                 outp);
}
